// FP8GEMM_77403900609091
// MI455X (gfx1250) — 8-bit matrix operand order repaired and checked by reading, compiled, not run on hardware
//
#include <hip/hip_runtime.h>
#include <hip/hip_bf16.h>

typedef __attribute__((ext_vector_type(16))) int   v16i;
typedef __attribute__((ext_vector_type(8)))  float v8f;
typedef __attribute__((ext_vector_type(4)))  int   v4i_t;

#define MAT_N     4096
#define TILE_K    128
#define BM        128
#define BN        128
#define LDS_PITCH 144
#define ABYTES    (BM * LDS_PITCH)
#define BBYTES    (BN * LDS_PITCH)
#define BUF_BYTES (ABYTES + BBYTES)

#define E4M3_MAX 448.0f

#if __has_builtin(__builtin_amdgcn_cvt_pk_fp8_f32)
__device__ __forceinline__ int pack4_e4m3(float a, float b, float c, float d) {
    a = fminf(fmaxf(a, -E4M3_MAX), E4M3_MAX);
    b = fminf(fmaxf(b, -E4M3_MAX), E4M3_MAX);
    c = fminf(fmaxf(c, -E4M3_MAX), E4M3_MAX);
    d = fminf(fmaxf(d, -E4M3_MAX), E4M3_MAX);
    int v = 0;
    v = __builtin_amdgcn_cvt_pk_fp8_f32(a, b, v, false);
    v = __builtin_amdgcn_cvt_pk_fp8_f32(c, d, v, true);
    return v;
}
#else
__device__ __forceinline__ unsigned int cvt1_e4m3(float x) {
    float ax = fabsf(x);
    ax = fminf(ax, E4M3_MAX);
    unsigned int s = (__float_as_uint(x) >> 31) << 7;
    if (ax < 0x1p-10f) return s;
    unsigned int u = __float_as_uint(ax);
    int e = (int)(u >> 23) - 127;
    if (e < -6) {
        int q = (int)rintf(ax * 512.0f);
        return s | (unsigned int)q;
    }
    unsigned int m    = u & 0x7FFFFFu;
    unsigned int keep = m >> 20;
    unsigned int rest = m & 0xFFFFFu;
    unsigned int rnd  = (rest > 0x80000u) || (rest == 0x80000u && (keep & 1u));
    keep += rnd;
    if (keep == 8u) { keep = 0u; e += 1; }
    if (e > 8) { e = 8; keep = 6u; }
    return s | ((unsigned int)(e + 7) << 3) | keep;
}
__device__ __forceinline__ int pack4_e4m3(float a, float b, float c, float d) {
    return (int)(cvt1_e4m3(a) | (cvt1_e4m3(b) << 8) |
                 (cvt1_e4m3(c) << 16) | (cvt1_e4m3(d) << 24));
}
#endif

#if __has_builtin(__builtin_amdgcn_global_load_async_to_lds_b128) && \
    __has_builtin(__builtin_amdgcn_s_wait_asynccnt)
#define USE_ASYNC_LDS 1
typedef __attribute__((address_space(1))) v4i_t* gvec_p;
typedef __attribute__((address_space(3))) v4i_t* lvec_p;
__device__ __forceinline__ void async_copy16(const unsigned char* g, unsigned char* l) {
    __builtin_amdgcn_global_load_async_to_lds_b128(
        (gvec_p)(unsigned long long)g,
        (lvec_p)(unsigned int)(unsigned long long)l, 0, 0);
}
#else
#define USE_ASYNC_LDS 0
#endif

__global__ __launch_bounds__(256) void quant_rows_kernel(
    const float* __restrict__ in, const float* __restrict__ scale,
    unsigned char* __restrict__ out8)
{
    const int row = blockIdx.x;
    const int t   = threadIdx.x;
    const float s = scale[row];
    const float* rin = in + (size_t)row * MAT_N;
    int* rout = (int*)(out8 + (size_t)row * MAT_N);

#pragma unroll
    for (int i = 0; i < 4; ++i) {
        const float4 f = *(const float4*)(rin + 4 * t + 1024 * i);
        rout[t + 256 * i] = pack4_e4m3(f.x * s, f.y * s, f.z * s, f.w * s);
    }
}

__global__ __launch_bounds__(256) void fp8_gemm_kernel(
    const unsigned char* __restrict__ A8,
    const unsigned char* __restrict__ B8,
    const float* __restrict__ sinv1,
    const float* __restrict__ sinv2,
    __hip_bfloat16* __restrict__ out)
{
    extern __shared__ unsigned char smem[];

    const int tid   = threadIdx.x;
    const int lane  = tid & 31;
    const int wave  = tid >> 5;
    const int wm    = (wave & 1) * 64;
    const int wn    = (wave >> 1) * 32;
    const int m0    = blockIdx.y * BM;
    const int n0    = blockIdx.x * BN;
    const int l16   = lane & 15;
    const int lhalf = lane >> 4;

    const int srow = tid >> 3;
    const int scol = (tid & 7) * 16;

    const unsigned char* gA = A8 + (size_t)(m0 + srow) * MAT_N + scol;
    const unsigned char* gB = B8 + (size_t)(n0 + srow) * MAT_N + scol;

    v8f acc[4][2];
#pragma unroll
    for (int i = 0; i < 4; ++i)
#pragma unroll
        for (int j = 0; j < 2; ++j)
            acc[i][j] = (v8f){0.f, 0.f, 0.f, 0.f, 0.f, 0.f, 0.f, 0.f};

    auto stage = [&](int buf, int kt) {
        unsigned char* dA = smem + buf * BUF_BYTES + srow * LDS_PITCH + scol;
        unsigned char* dB = dA + ABYTES;
#if USE_ASYNC_LDS
#pragma unroll
        for (int p = 0; p < 4; ++p)
            async_copy16(gA + (size_t)p * 32 * MAT_N + kt, dA + p * 32 * LDS_PITCH);
#pragma unroll
        for (int p = 0; p < 4; ++p)
            async_copy16(gB + (size_t)p * 32 * MAT_N + kt, dB + p * 32 * LDS_PITCH);
#else
#pragma unroll
        for (int p = 0; p < 4; ++p)
            *(int4*)(dA + p * 32 * LDS_PITCH) =
                *(const int4*)(gA + (size_t)p * 32 * MAT_N + kt);
#pragma unroll
        for (int p = 0; p < 4; ++p)
            *(int4*)(dB + p * 32 * LDS_PITCH) =
                *(const int4*)(gB + (size_t)p * 32 * MAT_N + kt);
#endif
    };

    stage(0, 0);
#if USE_ASYNC_LDS
    __builtin_amdgcn_s_wait_asynccnt(0);
#endif
    __syncthreads();

    int cb = 0;
    for (int kt = 0; kt < MAT_N; kt += TILE_K) {
        if (kt + TILE_K < MAT_N) stage(cb ^ 1, kt + TILE_K);

        const unsigned char* lA = smem + cb * BUF_BYTES;
        const unsigned char* lB = lA + ABYTES;

        v16i afrag[4];
#pragma unroll
        for (int i = 0; i < 4; ++i) {
            const unsigned char* base = lA + (wm + i * 16 + l16) * LDS_PITCH;
            unsigned long long* dst = (unsigned long long*)&afrag[i];
#pragma unroll
            for (int j = 0; j < 4; ++j)
                dst[j] = *(const unsigned long long*)(base + lhalf * 16 + (j >> 1) * 32 + (j & 1) * 8);
#pragma unroll
            for (int j = 0; j < 4; ++j)
                dst[4 + j] = *(const unsigned long long*)(base + 64 + lhalf * 16 + (j >> 1) * 32 + (j & 1) * 8);
        }
        v16i bfrag[2];
#pragma unroll
        for (int j = 0; j < 2; ++j) {
            const unsigned char* base = lB + (wn + j * 16 + l16) * LDS_PITCH;
            int4* dst = (int4*)&bfrag[j];
#pragma unroll
            for (int q = 0; q < 4; ++q)
                dst[q] = *(const int4*)(base + lhalf * 16 + q * 32);
        }

#pragma unroll
        for (int i = 0; i < 4; ++i)
#pragma unroll
            for (int j = 0; j < 2; ++j)
                acc[i][j] = __builtin_amdgcn_wmma_f32_16x16x128_fp8_fp8(
                    afrag[i], bfrag[j], (short)0, acc[i][j], false, false);

#if USE_ASYNC_LDS
        __builtin_amdgcn_s_wait_asynccnt(0);
#endif
        __syncthreads();
        cb ^= 1;
    }

#pragma unroll
    for (int i = 0; i < 4; ++i) {
#pragma unroll
        for (int j = 0; j < 2; ++j) {
            const int n = n0 + wn + j * 16 + l16;
            const float s2 = sinv2[n];
#pragma unroll
            for (int r = 0; r < 8; ++r) {
                const int m = m0 + wm + i * 16 + lhalf * 8 + r;
                const float v = acc[i][j][r] * sinv1[m] * s2;
                out[(size_t)m * MAT_N + n] = __float2bfloat16(v);
            }
        }
    }
}

extern "C" void kernel_launch(void* const* d_in, const int* in_sizes, int n_in,
                              void* d_out, int out_size, void* d_ws, size_t ws_size,
                              hipStream_t stream) {
    const float* x1     = (const float*)d_in[0];
    const float* x2     = (const float*)d_in[1];
    const float* scale1 = (const float*)d_in[2];
    const float* scale2 = (const float*)d_in[3];
    const float* sinv1  = (const float*)d_in[4];
    const float* sinv2  = (const float*)d_in[5];

    unsigned char* A8 = (unsigned char*)d_ws;
    unsigned char* B8 = A8 + (size_t)MAT_N * MAT_N;

    quant_rows_kernel<<<MAT_N, 256, 0, stream>>>(x1, scale1, A8);
    quant_rows_kernel<<<MAT_N, 256, 0, stream>>>(x2, scale2, B8);

    dim3 grid(MAT_N / BN, MAT_N / BM);
    fp8_gemm_kernel<<<grid, 256, 2 * BUF_BYTES, stream>>>(
        A8, B8, sinv1, sinv2, (__hip_bfloat16*)d_out);
}
